// AttnStyleUNet_37821482008657
// MI455X (gfx1250) — matrix operand order repaired and checked by reading, not compiled, not run on hardware
//
#include <hip/hip_runtime.h>
#include <hip/hip_bf16.h>
#include <cmath>
#include <cstdint>

typedef __attribute__((ext_vector_type(16))) _Float16 v16h;
typedef __attribute__((ext_vector_type(8)))  _Float16 v8h;
typedef __attribute__((ext_vector_type(8)))  float    v8f;
typedef __attribute__((ext_vector_type(4)))  unsigned int u32x4;
typedef __attribute__((ext_vector_type(8)))  int i32x8;
typedef __attribute__((ext_vector_type(4)))  int i32x4;

#define WMMA_F32_F16(a, b, c) \
  __builtin_amdgcn_wmma_f32_16x16x32_f16(false, (a), false, (b), (short)0, (c), false, false)

// ---------------------------------------------------------------------------
// TDM 2-D tile load: global (row-major, f16) -> LDS (contiguous tile).
//   row_elems   = tensor_dim0 = elements per tensor row (also dim0 stride)
//   nrows_tensor= tensor_dim1 (for OOB; tiles here are fully in-bounds)
//   tile_d0     = tile width in elements, tile_d1 = tile height (rows)
// Descriptor per CDNA5 ISA 8.3/8.4; groups 2/3 zero (2-D tile).
// This toolchain exposes the 6-arg builtin (extra zeroed i32x8 group).
// ---------------------------------------------------------------------------
__device__ __forceinline__ void tdm_load_tile_2d(unsigned lds_byte_off,
                                                 const _Float16* gptr,
                                                 unsigned row_elems,
                                                 unsigned nrows_tensor,
                                                 unsigned tile_d0,
                                                 unsigned tile_d1) {
  unsigned long long ga = (unsigned long long)(uintptr_t)gptr;
  u32x4 g0;
  g0[0] = 1u;                                            // count=1, user mode
  g0[1] = lds_byte_off;                                  // lds_addr (bytes)
  g0[2] = (unsigned)(ga & 0xffffffffull);                // global_addr[31:0]
  g0[3] = (unsigned)((ga >> 32) & 0x01ffffffull)         // global_addr[56:32]
          | (2u << 30);                                  // type = 2 ("image")
  i32x8 g1;
  g1[0] = (int)(1u << 16);                               // data_size=1 -> 2 bytes
  g1[1] = (int)((row_elems & 0xffffu) << 16);            // tensor_dim0[15:0]
  g1[2] = (int)((row_elems >> 16) | ((nrows_tensor & 0xffffu) << 16));
  g1[3] = (int)((nrows_tensor >> 16) | (tile_d0 << 16)); // tile_dim0
  g1[4] = (int)(tile_d1 & 0xffffu);                      // tile_dim1 (tile_dim2=0)
  g1[5] = (int)row_elems;                                // tensor_dim0_stride[31:0]
  g1[6] = 0;                                             // stride0 hi / stride1 lo
  g1[7] = 0;
  i32x4 gz4 = {0, 0, 0, 0};
  i32x8 gz8 = {0, 0, 0, 0, 0, 0, 0, 0};
  __builtin_amdgcn_tensor_load_to_lds(g0, g1, gz4, gz4, gz8, 0);
}

// ---------------------------------------------------------------------------
// Elementwise helpers
// ---------------------------------------------------------------------------
__global__ void k_f32_to_f16(const float* __restrict__ in, _Float16* __restrict__ out, int n) {
  int i = blockIdx.x * blockDim.x + threadIdx.x;
  if (i < n) out[i] = (_Float16)in[i];
}

// (B,C,N) f32  ->  (B,N,C) f16
__global__ void k_transpose_cn(const float* __restrict__ in, _Float16* __restrict__ out,
                               int C, int N, int total) {
  int i = blockIdx.x * blockDim.x + threadIdx.x;
  if (i >= total) return;
  int n = i % N;
  int t = i / N;
  int c = t % C;
  int b = t / C;
  out[((size_t)b * N + n) * C + c] = (_Float16)in[i];
}

// ---------------------------------------------------------------------------
// Direct Conv2d k=4 s=2 p=1 + ReLU  (small FLOP count -> scalar fp32 is fine)
// ---------------------------------------------------------------------------
__global__ void k_conv4s2(const float* __restrict__ in, const float* __restrict__ w,
                          const float* __restrict__ bias, float* __restrict__ out,
                          int Cin, int Cout, int Hin, int Win, int total) {
  int i = blockIdx.x * blockDim.x + threadIdx.x;
  if (i >= total) return;
  int Ho = Hin >> 1, Wo = Win >> 1;
  int x = i % Wo;
  int t = i / Wo;
  int y = t % Ho;  t /= Ho;
  int co = t % Cout;
  int b  = t / Cout;
  float acc = bias[co];
  for (int ci = 0; ci < Cin; ++ci) {
    const float* ip = in + ((size_t)(b * Cin + ci) * Hin) * Win;
    const float* wp = w + (size_t)(co * Cin + ci) * 16;
#pragma unroll
    for (int ky = 0; ky < 4; ++ky) {
      int iy = 2 * y + ky - 1;
      if ((unsigned)iy >= (unsigned)Hin) continue;
#pragma unroll
      for (int kx = 0; kx < 4; ++kx) {
        int ix = 2 * x + kx - 1;
        if ((unsigned)ix >= (unsigned)Win) continue;
        acc += ip[iy * Win + ix] * wp[ky * 4 + kx];
      }
    }
  }
  out[i] = acc > 0.f ? acc : 0.f;
}

// ---------------------------------------------------------------------------
// Direct ConvTranspose2d k=4 s=2 p=1, mode: 1=ReLU, 2=sigmoid
// weight layout (Cin, Cout, 4, 4) (torch ConvTranspose convention)
// ---------------------------------------------------------------------------
__global__ void k_deconv4s2(const float* __restrict__ in, const float* __restrict__ w,
                            const float* __restrict__ bias, float* __restrict__ out,
                            int Cin, int Cout, int Hin, int Win, int mode, int total) {
  int i = blockIdx.x * blockDim.x + threadIdx.x;
  if (i >= total) return;
  int Ho = Hin * 2, Wo = Win * 2;
  int x = i % Wo;
  int t = i / Wo;
  int y = t % Ho;  t /= Ho;
  int co = t % Cout;
  int b  = t / Cout;
  float acc = bias[co];
  int py = (y + 1) & 1;
  int px = (x + 1) & 1;
  for (int ci = 0; ci < Cin; ++ci) {
    const float* ip = in + ((size_t)(b * Cin + ci) * Hin) * Win;
    const float* wp = w + (size_t)(ci * Cout + co) * 16;
#pragma unroll
    for (int ty = 0; ty < 2; ++ty) {
      int ky = py + 2 * ty;
      int iy = (y + 1 - ky) >> 1;
      if ((unsigned)iy >= (unsigned)Hin) continue;
#pragma unroll
      for (int tx = 0; tx < 2; ++tx) {
        int kx = px + 2 * tx;
        int ix = (x + 1 - kx) >> 1;
        if ((unsigned)ix >= (unsigned)Win) continue;
        acc += ip[iy * Win + ix] * wp[ky * 4 + kx];
      }
    }
  }
  if (mode == 1)      acc = acc > 0.f ? acc : 0.f;
  else if (mode == 2) acc = 1.f / (1.f + __expf(-acc));
  out[i] = acc;
}

// ---------------------------------------------------------------------------
// Projection GEMM via WMMA:  out[b,n,d] = sum_c Xt[b,n,c] * W[d,c] + bias[d]
//   Xt: (B,N,C) f16 row-major; W: (C,C) f16 row-major (d, c) = torch (out,in)
//   storeT==0 -> out row-major (B,N,C) f16   (Q, K)
//   storeT==1 -> out transposed (B,C,N) f16  (V^T)
// ---------------------------------------------------------------------------
__global__ void k_proj_gemm(const _Float16* __restrict__ Xt, const _Float16* __restrict__ W,
                            const float* __restrict__ bias, _Float16* __restrict__ out,
                            int N, int C, int storeT) {
  int lane = threadIdx.x & 31;
  int wv   = threadIdx.x >> 5;
  int nl   = lane & 15;
  int g    = lane >> 4;
  int n0   = (blockIdx.x * 4 + wv) * 16;
  int d0   = blockIdx.y * 16;
  int b    = blockIdx.z;

  const _Float16* xrow = Xt + ((size_t)b * N + n0 + nl) * C;
  const _Float16* wrow = W + (size_t)(d0 + nl) * C;

  v8f acc = {};
  for (int c0 = 0; c0 < C; c0 += 32) {
    v8h alo = *(const v8h*)(xrow + c0 + 8 * g);
    v8h ahi = *(const v8h*)(xrow + c0 + 16 + 8 * g);
    v8h blo = *(const v8h*)(wrow + c0 + 8 * g);
    v8h bhi = *(const v8h*)(wrow + c0 + 16 + 8 * g);
    v16h af, bf;
#pragma unroll
    for (int e = 0; e < 8; ++e) {
      af[e] = alo[e]; af[e + 8] = ahi[e];
      bf[e] = blo[e]; bf[e + 8] = bhi[e];
    }
    acc = WMMA_F32_F16(af, bf, acc);
  }

  float bv = bias[d0 + nl];
  if (storeT) {
    v8h st;
#pragma unroll
    for (int r = 0; r < 8; ++r) st[r] = (_Float16)(acc[r] + bv);
    *(v8h*)(out + ((size_t)b * C + d0 + nl) * N + n0 + 8 * g) = st;
  } else {
#pragma unroll
    for (int r = 0; r < 8; ++r)
      out[((size_t)b * N + n0 + r + 8 * g) * C + d0 + nl] = (_Float16)(acc[r] + bv);
  }
}

// ---------------------------------------------------------------------------
// Flash attention, wave-level WMMA f16, K/V tiles staged in LDS by the TDM.
//   Q,K: (B,N,C) f16 row-major; Vt: (B,C,N) f16; out: (B,C,N) f32
// Block = 128 threads = 4 waves sharing one (K,V) tile per iteration:
//   wave 0 issues two tensor_load_to_lds (32xC K tile, Cx32 V tile),
//   waits TENSORcnt, then all waves consume fragments from LDS.
// S^T-tile trick: D-layout of S^T matches the A-fragment layout for P x V.
// ---------------------------------------------------------------------------
template <int C>
__global__ void k_flash_attn(const _Float16* __restrict__ Q, const _Float16* __restrict__ K,
                             const _Float16* __restrict__ Vt, float* __restrict__ out,
                             int N, float scale) {
  constexpr int KC = C / 32;   // d-chunks for S = Q.K^T
  constexpr int NT = C / 16;   // output d-tiles
  __shared__ __align__(16) _Float16 kbuf[32 * C];  // 32 m-rows x C
  __shared__ __align__(16) _Float16 vbuf[C * 32];  // C d-rows x 32 m-cols

  int lane = threadIdx.x & 31;
  int wv   = threadIdx.x >> 5;
  int nl   = lane & 15;
  int g    = lane >> 4;
  int n0   = (blockIdx.x * 4 + wv) * 16;
  int b    = blockIdx.y;

  const _Float16* Qb = Q + (size_t)b * N * C;
  const _Float16* Kb = K + (size_t)b * N * C;
  const _Float16* Vb = Vt + (size_t)b * C * N;
  unsigned k_lds = (unsigned)(uintptr_t)(&kbuf[0]);
  unsigned v_lds = (unsigned)(uintptr_t)(&vbuf[0]);

  // Preload Q as B-fragments: lane col n = n0+nl, element e<8 -> d = kc*32+8g+e, e>=8 -> d = kc*32+16+8g+(e-8).
  v16h qf[KC];
#pragma unroll
  for (int kc = 0; kc < KC; ++kc) {
    const _Float16* p = Qb + (size_t)(n0 + nl) * C + kc * 32 + 8 * g;
    v8h lo = *(const v8h*)p;
    v8h hi = *(const v8h*)(p + 16);
#pragma unroll
    for (int e = 0; e < 8; ++e) { qf[kc][e] = lo[e]; qf[kc][e + 8] = hi[e]; }
  }

  v8f zero = {};
  v8f o[NT];
#pragma unroll
  for (int t = 0; t < NT; ++t) o[t] = zero;
  float mrun = -1e30f, lrun = 0.f;

  for (int m0 = 0; m0 < N; m0 += 32) {
    __syncthreads();                    // previous tile fully consumed
    if (wv == 0) {
      // K tile: rows [m0, m0+32) x C  -> kbuf (contiguous)
      tdm_load_tile_2d(k_lds, Kb + (size_t)m0 * C, (unsigned)C, (unsigned)N,
                       (unsigned)C, 32u);
      // V tile: C rows x cols [m0, m0+32) -> vbuf (contiguous)
      tdm_load_tile_2d(v_lds, Vb + m0, (unsigned)N, (unsigned)C,
                       32u, (unsigned)C);
      __builtin_amdgcn_s_wait_tensorcnt(0);
    }
    __syncthreads();                    // tile visible to all waves

    // --- two S^T tiles (m rows x n cols) from LDS K ---
    v8f s0 = {}, s1 = {};
#pragma unroll
    for (int kc = 0; kc < KC; ++kc) {
      const _Float16* kp0 = &kbuf[(size_t)nl * C + kc * 32];
      const _Float16* kp1 = &kbuf[(size_t)(16 + nl) * C + kc * 32];
      v8h a0l = *(const v8h*)(kp0 + 8 * g);
      v8h a0h = *(const v8h*)(kp0 + 16 + 8 * g);
      v8h a1l = *(const v8h*)(kp1 + 8 * g);
      v8h a1h = *(const v8h*)(kp1 + 16 + 8 * g);
      v16h af0, af1;
#pragma unroll
      for (int e = 0; e < 8; ++e) {
        af0[e] = a0l[e]; af0[e + 8] = a0h[e];
        af1[e] = a1l[e]; af1[e + 8] = a1h[e];
      }
      s0 = WMMA_F32_F16(af0, qf[kc], s0);
      s1 = WMMA_F32_F16(af1, qf[kc], s1);
    }

    // --- online softmax over m (per lane: n = nl; lanes nl / nl+16 pair) ---
    float tmax = -1e30f;
#pragma unroll
    for (int r = 0; r < 8; ++r) {
      s0[r] *= scale; s1[r] *= scale;
      tmax = fmaxf(tmax, fmaxf(s0[r], s1[r]));
    }
    tmax = fmaxf(tmax, __shfl_xor(tmax, 16, 32));
    float mnew  = fmaxf(mrun, tmax);
    float alpha = __expf(mrun - mnew);
    float p0[8], p1[8], psum = 0.f;
#pragma unroll
    for (int r = 0; r < 8; ++r) {
      p0[r] = __expf(s0[r] - mnew);
      p1[r] = __expf(s1[r] - mnew);
      psum += p0[r] + p1[r];
    }
    psum += __shfl_xor(psum, 16, 32);
    lrun = lrun * alpha + psum;
    mrun = mnew;

#pragma unroll
    for (int r = 0; r < 8; ++r) {
      float ar = __shfl(alpha, r + 8 * g, 32);
#pragma unroll
      for (int t = 0; t < NT; ++t) o[t][r] *= ar;
    }

    // P (16n x 32m) as A-fragment: in-lane repack of the two S^T D-tiles
    v16h pf;
#pragma unroll
    for (int e = 0; e < 8; ++e) {
      pf[e]     = (_Float16)p0[e];
      pf[e + 8] = (_Float16)p1[e];
    }

    // O += P x V from LDS V: lane col d = t*16+nl, element e<8 -> m = 8g+e, e>=8 -> m = 16+8g+(e-8)
#pragma unroll
    for (int t = 0; t < NT; ++t) {
      const _Float16* vp = &vbuf[(size_t)(t * 16 + nl) * 32 + 8 * g];
      v8h vl = *(const v8h*)vp;
      v8h vh = *(const v8h*)(vp + 16);
      v16h vf;
#pragma unroll
      for (int e = 0; e < 8; ++e) { vf[e] = vl[e]; vf[e + 8] = vh[e]; }
      o[t] = WMMA_F32_F16(pf, vf, o[t]);
    }
  }

  // epilogue: out[b, d, n] = O[n, d] / l[n]
  float linv = 1.f / lrun;
  float* ob = out + (size_t)b * C * N;
#pragma unroll
  for (int r = 0; r < 8; ++r) {
    float lr = __shfl(linv, r + 8 * g, 32);
#pragma unroll
    for (int t = 0; t < NT; ++t)
      ob[(size_t)(t * 16 + nl) * N + n0 + 8 * g + r] = o[t][r] * lr;
  }
}

// ---------------------------------------------------------------------------
// Host-side orchestration
// ---------------------------------------------------------------------------
static inline int cdiv(int a, int b) { return (a + b - 1) / b; }

struct AttnWS {
  _Float16 *Xt, *Yt, *Wq, *Wk, *Wv, *Qh, *Kh, *Vt;
};

static void run_attn(const float* X, const float* Y, int C, int N,
                     const float* qw, const float* qb,
                     const float* kw, const float* kb,
                     const float* vw, const float* vb,
                     float* aout, const AttnWS& ws, hipStream_t stream) {
  const int B = 4;
  int tot = B * C * N;
  k_transpose_cn<<<cdiv(tot, 256), 256, 0, stream>>>(X, ws.Xt, C, N, tot);
  const _Float16* Yt = ws.Xt;
  if (Y != X) {
    k_transpose_cn<<<cdiv(tot, 256), 256, 0, stream>>>(Y, ws.Yt, C, N, tot);
    Yt = ws.Yt;
  }
  int wn = C * C;
  k_f32_to_f16<<<cdiv(wn, 256), 256, 0, stream>>>(qw, ws.Wq, wn);
  k_f32_to_f16<<<cdiv(wn, 256), 256, 0, stream>>>(kw, ws.Wk, wn);
  k_f32_to_f16<<<cdiv(wn, 256), 256, 0, stream>>>(vw, ws.Wv, wn);

  dim3 pg(N / 64, C / 16, B);
  k_proj_gemm<<<pg, 128, 0, stream>>>(ws.Xt, ws.Wq, qb, ws.Qh, N, C, 0);
  k_proj_gemm<<<pg, 128, 0, stream>>>(Yt,    ws.Wk, kb, ws.Kh, N, C, 0);
  k_proj_gemm<<<pg, 128, 0, stream>>>(Yt,    ws.Wv, vb, ws.Vt, N, C, 1);

  float scale = 1.f / sqrtf((float)C);
  dim3 fg(N / 64, B);
  if (C == 64)
    k_flash_attn<64><<<fg, 128, 0, stream>>>(ws.Qh, ws.Kh, ws.Vt, aout, N, scale);
  else if (C == 128)
    k_flash_attn<128><<<fg, 128, 0, stream>>>(ws.Qh, ws.Kh, ws.Vt, aout, N, scale);
  else
    k_flash_attn<256><<<fg, 128, 0, stream>>>(ws.Qh, ws.Kh, ws.Vt, aout, N, scale);
}

extern "C" void kernel_launch(void* const* d_in, const int* in_sizes, int n_in,
                              void* d_out, int out_size, void* d_ws, size_t ws_size,
                              hipStream_t stream) {
  (void)in_sizes; (void)n_in; (void)out_size; (void)ws_size;
  const float* content = (const float*)d_in[0];
  const float* enc1_w = (const float*)d_in[1];  const float* enc1_b = (const float*)d_in[2];
  const float* enc2_w = (const float*)d_in[3];  const float* enc2_b = (const float*)d_in[4];
  const float* enc3_w = (const float*)d_in[5];  const float* enc3_b = (const float*)d_in[6];
  const float* dec1_w = (const float*)d_in[7];  const float* dec1_b = (const float*)d_in[8];
  const float* dec2_w = (const float*)d_in[9];  const float* dec2_b = (const float*)d_in[10];
  const float* dec3_w = (const float*)d_in[11]; const float* dec3_b = (const float*)d_in[12];
  const float* ca1_qw = (const float*)d_in[13]; const float* ca1_qb = (const float*)d_in[14];
  const float* ca1_kw = (const float*)d_in[15]; const float* ca1_kb = (const float*)d_in[16];
  const float* ca1_vw = (const float*)d_in[17]; const float* ca1_vb = (const float*)d_in[18];
  const float* ca2_qw = (const float*)d_in[19]; const float* ca2_qb = (const float*)d_in[20];
  const float* ca2_kw = (const float*)d_in[21]; const float* ca2_kb = (const float*)d_in[22];
  const float* ca2_vw = (const float*)d_in[23]; const float* ca2_vb = (const float*)d_in[24];
  const float* ca3_qw = (const float*)d_in[25]; const float* ca3_qb = (const float*)d_in[26];
  const float* ca3_kw = (const float*)d_in[27]; const float* ca3_kb = (const float*)d_in[28];
  const float* ca3_vw = (const float*)d_in[29]; const float* ca3_vb = (const float*)d_in[30];

  float* out = (float*)d_out;
  // d_out layout: out(196608) | e1(1048576) | e2(524288) | e3(262144) | c1 | c2 | c3
  float* e1 = out + 196608;
  float* e2 = out + 1245184;
  float* e3 = out + 1769472;

  char* p = (char*)d_ws;
  auto carve = [&](size_t bytes) -> void* {
    void* r = (void*)p;
    p += (bytes + 255) & ~(size_t)255;
    return r;
  };
  float* a3 = (float*)carve((size_t)262144 * 4);
  float* d1 = (float*)carve((size_t)524288 * 4);
  float* a2 = (float*)carve((size_t)524288 * 4);
  float* d2 = (float*)carve((size_t)1048576 * 4);
  float* a1 = (float*)carve((size_t)1048576 * 4);
  AttnWS ws;
  ws.Xt = (_Float16*)carve((size_t)1048576 * 2);
  ws.Yt = (_Float16*)carve((size_t)1048576 * 2);
  ws.Wq = (_Float16*)carve((size_t)65536 * 2);
  ws.Wk = (_Float16*)carve((size_t)65536 * 2);
  ws.Wv = (_Float16*)carve((size_t)65536 * 2);
  ws.Qh = (_Float16*)carve((size_t)1048576 * 2);
  ws.Kh = (_Float16*)carve((size_t)1048576 * 2);
  ws.Vt = (_Float16*)carve((size_t)1048576 * 2);

  // ---------------- encoder ----------------
  { int tot = 4 * 64 * 64 * 64;
    k_conv4s2<<<cdiv(tot, 256), 256, 0, stream>>>(content, enc1_w, enc1_b, e1, 3, 64, 128, 128, tot); }
  { int tot = 4 * 128 * 32 * 32;
    k_conv4s2<<<cdiv(tot, 256), 256, 0, stream>>>(e1, enc2_w, enc2_b, e2, 64, 128, 64, 64, tot); }
  { int tot = 4 * 256 * 16 * 16;
    k_conv4s2<<<cdiv(tot, 256), 256, 0, stream>>>(e2, enc3_w, enc3_b, e3, 128, 256, 32, 32, tot); }

  // ---------------- decoder with cross-attention ----------------
  run_attn(e3, e3, 256, 256, ca1_qw, ca1_qb, ca1_kw, ca1_kb, ca1_vw, ca1_vb, a3, ws, stream);
  { int tot = 4 * 128 * 32 * 32;
    k_deconv4s2<<<cdiv(tot, 256), 256, 0, stream>>>(a3, dec1_w, dec1_b, d1, 256, 128, 16, 16, 1, tot); }

  run_attn(d1, e2, 128, 1024, ca2_qw, ca2_qb, ca2_kw, ca2_kb, ca2_vw, ca2_vb, a2, ws, stream);
  { int tot = 4 * 64 * 64 * 64;
    k_deconv4s2<<<cdiv(tot, 256), 256, 0, stream>>>(a2, dec2_w, dec2_b, d2, 128, 64, 32, 32, 1, tot); }

  run_attn(d2, e1, 64, 4096, ca3_qw, ca3_qb, ca3_kw, ca3_kb, ca3_vw, ca3_vb, a1, ws, stream);
  { int tot = 4 * 3 * 128 * 128;
    k_deconv4s2<<<cdiv(tot, 256), 256, 0, stream>>>(a1, dec3_w, dec3_b, out, 64, 3, 64, 64, 2, tot); }

  // (c1,c2,c3) == (e1,e2,e3): duplicate the contiguous feature region
  (void)hipMemcpyAsync(out + 2031616, e1, (size_t)1835008 * sizeof(float),
                       hipMemcpyDeviceToDevice, stream);
}
